// ObCrossTrEncoder_65678639891296
// MI455X (gfx1250) — matrix operand order repaired and checked by reading, not compiled, not run on hardware
//
#include <hip/hip_runtime.h>
#include <hip/hip_bf16.h>

// ---------------- constants ----------------
#define NC    9000
#define HGRID 30
#define WGRID 60
#define HW    1800
#define NOBS  27000
#define DIM   256
#define LMAX  12
#define LPAD  16
#define NBG   8
#define DEPTH 8
#define FF    1024
#define CL    512
#define NBGDIM 2048   // NBG*DIM

typedef __attribute__((ext_vector_type(16))) _Float16 v16h;
typedef __attribute__((ext_vector_type(8)))  _Float16 v8h;
typedef __attribute__((ext_vector_type(8)))  float    v8f;

// ---------------- WMMA helpers (f16 in, f32 acc, 16x16x32) ----------------
__device__ __forceinline__ v8f wmma32(v16h a, v16h b, v8f c) {
    return __builtin_amdgcn_wmma_f32_16x16x32_f16(
        /*neg_a=*/false, a, /*neg_b=*/false, b,
        /*c_mod=*/(short)0, c, /*reuse_a=*/false, /*reuse_b=*/false);
}

__device__ __forceinline__ v16h cat16(v8h lo, v8h hi) {
    return __builtin_shufflevector(lo, hi, 0, 1, 2, 3, 4, 5, 6, 7,
                                   8, 9, 10, 11, 12, 13, 14, 15);
}

// A fragment A[16][32] row-major (base -> A[0][k0], leading dim ld, 16B aligned).
// Contiguous K order used here (not the ISA table order): lane L: M=L%16; K = kg..kg+15, kg=(L/16)*16
// -> two contiguous 16-byte loads per lane.
__device__ __forceinline__ v16h loadAv(const _Float16* base, int ld, int lane) {
    const _Float16* p = base + (lane & 15) * ld + ((lane >> 4) << 4);
    v8h lo = *(const v8h*)p;
    v8h hi = *(const v8h*)(p + 8);
    return cat16(lo, hi);
}

// A fragment with K 16..31 structurally zero (attention weights, ld = 16)
__device__ __forceinline__ v16h loadA_k16v(const _Float16* base, int lane) {
    const _Float16* p = base + (lane & 15) * 16;
    if (lane < 16) return cat16(*(const v8h*)p, *(const v8h*)(p + 8));
    v8h z = {};
    return cat16(z, z);
}

// B fragment from pre-swizzled weights: per 32x16 (K,N) block the 32 lanes'
// 16 halves are stored contiguously -> two coalesced 16-byte global loads.
// Block order: blk = ntile*KT + ktile, KT = K/32.
__device__ __forceinline__ v16h loadBs(const _Float16* mat, int KT, int ktile,
                                       int ntile, int lane) {
    const _Float16* p = mat + (((size_t)(ntile * KT + ktile)) << 9) + lane * 16;
    v8h lo = *(const v8h*)p;
    v8h hi = *(const v8h*)(p + 8);
    return cat16(lo, hi);
}

// B^T fragment: logical B[K][N] = src[N*stride + K] (src row-major [N][K]).
// Per lane contiguous in K -> two 16-byte LDS loads.
__device__ __forceinline__ v16h loadBTv(const _Float16* src, int stride, int lane) {
    const _Float16* p = src + (lane & 15) * stride + ((lane >> 4) << 4);
    v8h lo = *(const v8h*)p;
    v8h hi = *(const v8h*)(p + 8);
    return cat16(lo, hi);
}

// B fragment for att@V from transposed V tile smVT[256 cols][16 slots]:
// B[K=slot][N=col] = smVT[col*16 + K]; K valid < 16, lanes 16..31 carry zeros.
__device__ __forceinline__ v16h loadBT_k16v(const _Float16* srcT, int lane) {
    if (lane < 16) {
        const _Float16* p = srcT + (lane & 15) * 16;
        v8h lo = *(const v8h*)p;
        v8h hi = *(const v8h*)(p + 8);
        return cat16(lo, hi);
    }
    v8h z = {};
    return cat16(z, z);
}

// Store f32 accumulator tile to f16 LDS row-major, leading dim 256 (scatter).
__device__ __forceinline__ void storeC256(v8f c, _Float16* dst, int n0, int lane) {
#pragma unroll
    for (int r = 0; r < 8; ++r) {
        int M = r + ((lane >> 4) << 3);
        dst[M * 256 + n0 + (lane & 15)] = (_Float16)c[r];
    }
}

// Store accumulator tile transposed into [col][slot] layout: the 8 values of a
// lane land on consecutive halves -> single packed 16-byte LDS store.
__device__ __forceinline__ void storeCT16(v8f c, _Float16* dstT, int n0, int lane) {
    v8h packed;
#pragma unroll
    for (int r = 0; r < 8; ++r) packed[r] = (_Float16)c[r];
    *(v8h*)(dstT + (n0 + (lane & 15)) * 16 + ((lane >> 4) << 3)) = packed;
}

__device__ __forceinline__ float geluf(float x) {
    float x3 = x * x * x;
    return 0.5f * x * (1.f + tanhf(0.7978845608028654f * (x + 0.044715f * x3)));
}

// LayerNorm one 256-row by one wave -> f16 out (with gain/bias)
__device__ __forceinline__ void ln_row(const float* row, const float* g, const float* b,
                                       _Float16* out, int lane) {
    float loc[8];
    float s = 0.f;
#pragma unroll
    for (int j = 0; j < 8; ++j) { loc[j] = row[lane * 8 + j]; s += loc[j]; }
#pragma unroll
    for (int off = 16; off > 0; off >>= 1) s += __shfl_xor(s, off, 32);
    float mean = s * (1.f / 256.f);
    float vv = 0.f;
#pragma unroll
    for (int j = 0; j < 8; ++j) { float d = loc[j] - mean; vv += d * d; }
#pragma unroll
    for (int off = 16; off > 0; off >>= 1) vv += __shfl_xor(vv, off, 32);
    float r = rsqrtf(vv * (1.f / 256.f) + 1e-5f);
    v8h packed;
#pragma unroll
    for (int j = 0; j < 8; ++j) {
        int c = lane * 8 + j;
        packed[j] = (_Float16)((loc[j] - mean) * r * g[c] + b[c]);
    }
    *(v8h*)(out + lane * 8) = packed;
}

// ---------------- kernel 1: bucket + normalize context rows ----------------
__global__ void bucket_norm_kernel(const float* __restrict__ x,
                                   const int* __restrict__ li,
                                   int* __restrict__ counts,
                                   _Float16* __restrict__ nctx) {
    int lane = threadIdx.x & 31;
    int wv   = threadIdx.x >> 5;
    int obs  = blockIdx.x * (blockDim.x >> 5) + wv;
    if (obs >= NOBS) return;
    int fi = li[obs * 3] * HW + li[obs * 3 + 1] * WGRID + li[obs * 3 + 2];
    int slot = 0;
    if (lane == 0) slot = atomicAdd(&counts[fi], 1);
    slot = __shfl(slot, 0, 32);
    const float* row = x + (size_t)obs * DIM;
    float loc[8];
    float s = 0.f;
#pragma unroll
    for (int j = 0; j < 8; ++j) { loc[j] = row[lane * 8 + j]; s += loc[j]; }
#pragma unroll
    for (int off = 16; off > 0; off >>= 1) s += __shfl_xor(s, off, 32);
    float mean = s * (1.f / 256.f);
    float vv = 0.f;
#pragma unroll
    for (int j = 0; j < 8; ++j) { float d = loc[j] - mean; vv += d * d; }
#pragma unroll
    for (int off = 16; off > 0; off >>= 1) vv += __shfl_xor(vv, off, 32);
    float r = rsqrtf(vv * (1.f / 256.f) + 1e-5f);
    if (slot < LMAX) {
        _Float16* dst = nctx + (size_t)fi * (LPAD * DIM) + slot * DIM;
        v8h packed;
#pragma unroll
        for (int j = 0; j < 8; ++j) packed[j] = (_Float16)((loc[j] - mean) * r);
        *(v8h*)(dst + lane * 8) = packed;
    }
}

// ---------------- kernel 2: f32 weights -> f16, pre-swizzled to B-fragment order --
// dst element i: blk = (i%per)/512 with blk = ntile*KT + ktile, r = i%512,
// lane = r/16, h = r%16; n = lane&15, k0 = (lane>>4)*16; source (K,N) element.
__global__ void swz_kernel(const float* __restrict__ src, _Float16* __restrict__ dst,
                           int K, int N, int batch) {
    int per = K * N;
    long long total = (long long)per * batch;
    long long i = (long long)blockIdx.x * blockDim.x + threadIdx.x;
    if (i >= total) return;
    int b = (int)(i / per), o = (int)(i % per);
    int blk = o >> 9, r = o & 511;
    int L = r >> 4, h = r & 15;
    int KT = K >> 5;
    int ktile = blk % KT, ntile = blk / KT;
    int n = L & 15, k0 = (L >> 4) << 4;
    int Ki = ktile * 32 + k0 + h;
    int Ni = ntile * 16 + n;
    dst[i] = (_Float16)src[(size_t)b * per + (size_t)Ki * N + Ni];
}

// ---------------- kernel 3: per-cell 8-layer cross transformer ----------------
__global__ __launch_bounds__(512, 1)
void obs_cross_tr_kernel(const int* __restrict__ counts,
                         const _Float16* __restrict__ nctx,
                         const float* __restrict__ bg,
                         const float* __restrict__ ln1g, const float* __restrict__ ln1b,
                         const float* __restrict__ lncg, const float* __restrict__ lncb,
                         const _Float16* __restrict__ WqH, const _Float16* __restrict__ WkH,
                         const _Float16* __restrict__ WvH, const _Float16* __restrict__ WoH,
                         const float* __restrict__ ln2g, const float* __restrict__ ln2b,
                         const _Float16* __restrict__ Wf1H, const float* __restrict__ bf1,
                         const _Float16* __restrict__ Wf2H, const float* __restrict__ bf2,
                         _Float16* __restrict__ tqH) {
    const int cell = blockIdx.x;
    const int tid = threadIdx.x, lane = tid & 31, wv = tid >> 5;

    __shared__ float    tq32[8 * 256];     // persistent fp32 residual stream
    __shared__ _Float16 smA[16 * 256];     // LN'd activations (rows 8-15 stay zero)
    __shared__ _Float16 smC[16 * 256];     // per-layer scaled context
    __shared__ _Float16 smAtt[8 * 16 * 16];// attention weights per head
    __shared__ _Float16 smBig[16 * 1024];  // Q|K|VT|AO during attention; FFN hidden later
    _Float16* smQ  = smBig;                // [16 tok][256]
    _Float16* smK  = smBig + 4096;         // [16 slot][256]
    _Float16* smVT = smBig + 8192;         // [256 col][16 slot] (transposed)
    _Float16* smAO = smBig + 12288;        // [16 tok][256]

    const _Float16* cellctx = nctx + (size_t)cell * (LPAD * DIM);
    int vc = counts[cell];
    vc = vc < 1 ? 1 : (vc > LMAX ? LMAX : vc);

    for (int i = tid; i < 8 * 256; i += 512) tq32[i] = bg[i];
    for (int i = tid + 8 * 256; i < 16 * 256; i += 512) smA[i] = (_Float16)0.f;
    __syncthreads();

    for (int L = 0; L < DEPTH; ++L) {
        // --- stage c = nctx * lnc_g + lnc_b (zero ctx rows -> bias, as reference) ---
        {
            const float* gc = lncg + L * 256;
            const float* bc = lncb + L * 256;
            for (int i = tid; i < 16 * 256; i += 512) {
                int col = i & 255;
                smC[i] = (_Float16)((float)cellctx[i] * gc[col] + bc[col]);
            }
        }
        if (wv < 8) ln_row(tq32 + wv * 256, ln1g + L * 256, ln1b + L * 256, smA + wv * 256, lane);
        __syncthreads();

        // --- Q/K/V projections: wave -> one 16-col N-tile, 8 K-steps of 32 ---
        {
            const _Float16* wq = WqH + (size_t)L * 65536;
            const _Float16* wk = WkH + (size_t)L * 65536;
            const _Float16* wvp = WvH + (size_t)L * 65536;
            v8f aq = {}, ak = {}, av = {};
            for (int ks = 0; ks < 8; ++ks) {
                v16h aA = loadAv(smA + ks * 32, 256, lane);
                v16h aC = loadAv(smC + ks * 32, 256, lane);
                v16h bq = loadBs(wq, 8, ks, wv, lane);
                v16h bk = loadBs(wk, 8, ks, wv, lane);
                v16h bv = loadBs(wvp, 8, ks, wv, lane);
                aq = wmma32(aA, bq, aq);
                ak = wmma32(aC, bk, ak);
                av = wmma32(aC, bv, av);
            }
            storeC256(aq, smQ, wv * 16, lane);
            storeC256(ak, smK, wv * 16, lane);
            storeCT16(av, smVT, wv * 16, lane);   // transposed, packed b128 store
        }
        __syncthreads();

        // --- scores + masked softmax (wave w = head w; K = dh = 32) ---
        if (wv < 8) {
            const int hd = wv;
            v16h a = loadAv(smQ + hd * 32, 256, lane);
            v16h b = loadBTv(smK + hd * 32, 256, lane);  // B[kk][slot] = K[slot][kk]
            v8f sc = {};
            sc = wmma32(a, b, sc);
            const int slot = lane & 15;
#pragma unroll
            for (int r = 0; r < 8; ++r) {
                float s = sc[r] * 0.17677669529663689f;  // 1/sqrt(32)
                if (slot >= vc) s = -1e30f;
                float m = s;
#pragma unroll
                for (int off = 8; off > 0; off >>= 1) m = fmaxf(m, __shfl_xor(m, off, 16));
                float e = __expf(s - m);
                float se = e;
#pragma unroll
                for (int off = 8; off > 0; off >>= 1) se += __shfl_xor(se, off, 16);
                int M = r + ((lane >> 4) << 3);
                smAtt[hd * 256 + M * 16 + slot] = (_Float16)((M < 8) ? (e / se) : 0.f);
            }
        }
        __syncthreads();

        // --- att @ V: 16 tasks = 8 heads x 2 col-halves, slot-K zero-padded to 32 ---
        {
            const int hd = wv >> 1, half = wv & 1;
            v16h a = loadA_k16v(smAtt + hd * 256, lane);
            v16h b = loadBT_k16v(smVT + (hd * 32 + half * 16) * 16, lane);
            v8f ao = {};
            ao = wmma32(a, b, ao);
            storeC256(ao, smAO, hd * 32 + half * 16, lane);
        }
        __syncthreads();

        // --- O projection + residual into tq32 ---
        {
            const _Float16* wo = WoH + (size_t)L * 65536;
            v8f acc = {};
            for (int ks = 0; ks < 8; ++ks) {
                v16h a = loadAv(smAO + ks * 32, 256, lane);
                v16h b = loadBs(wo, 8, ks, wv, lane);
                acc = wmma32(a, b, acc);
            }
#pragma unroll
            for (int r = 0; r < 8; ++r) {
                int M = r + ((lane >> 4) << 3);
                if (M < 8) tq32[M * 256 + wv * 16 + (lane & 15)] += acc[r];
            }
        }
        __syncthreads();

        // --- ln2 -> smA rows 0..7 ---
        if (wv < 8) ln_row(tq32 + wv * 256, ln2g + L * 256, ln2b + L * 256, smA + wv * 256, lane);
        __syncthreads();

        // --- FFN1 (256 -> 1024) + gelu, hidden in smBig ---
        {
            const _Float16* w1 = Wf1H + (size_t)L * 262144;
            const float* b1 = bf1 + L * 1024;
#pragma unroll
            for (int t = 0; t < 4; ++t) {
                const int nt = wv * 4 + t;
                v8f acc = {};
                for (int ks = 0; ks < 8; ++ks) {
                    v16h a = loadAv(smA + ks * 32, 256, lane);
                    v16h b = loadBs(w1, 8, ks, nt, lane);
                    acc = wmma32(a, b, acc);
                }
#pragma unroll
                for (int r = 0; r < 8; ++r) {
                    int M = r + ((lane >> 4) << 3);
                    int col = nt * 16 + (lane & 15);
                    smBig[M * 1024 + col] = (_Float16)geluf(acc[r] + b1[col]);
                }
            }
        }
        __syncthreads();

        // --- FFN2 (1024 -> 256) + bias + residual ---
        {
            const _Float16* w2 = Wf2H + (size_t)L * 262144;
            const float* b2 = bf2 + L * 256;
            v8f acc = {};
            for (int ks = 0; ks < 32; ++ks) {
                v16h a = loadAv(smBig + ks * 32, 1024, lane);
                v16h b = loadBs(w2, 32, ks, wv, lane);
                acc = wmma32(a, b, acc);
            }
#pragma unroll
            for (int r = 0; r < 8; ++r) {
                int M = r + ((lane >> 4) << 3);
                if (M < 8) tq32[M * 256 + wv * 16 + (lane & 15)] += acc[r] + b2[wv * 16 + (lane & 15)];
            }
        }
        __syncthreads();
    }

    // flush final tokens as f16 row [2048] for the batched output GEMM
    _Float16* dst = tqH + (size_t)cell * NBGDIM;
    for (int i = tid; i < NBGDIM; i += 512) dst[i] = (_Float16)tq32[i];
}

// ---------------- kernel 4: [9000 x 2048] @ [2048 x 512] + bias, gated by counts ----
__global__ __launch_bounds__(512, 1)
void final_proj_kernel(const _Float16* __restrict__ tqH,
                       const _Float16* __restrict__ WpH,
                       const float* __restrict__ bp,
                       const int* __restrict__ counts,
                       float* __restrict__ out) {
    __shared__ _Float16 sA[16 * 32];
    const int tid = threadIdx.x, lane = tid & 31, wv = tid >> 5;
    const int row0 = blockIdx.x * 16;
    v8f acc0 = {}, acc1 = {};
    for (int ks = 0; ks < 64; ++ks) {
        {   // stage 16x32 A tile (512 threads -> 1 elem each)
            int r = tid >> 5, kk = tid & 31;
            int cell = row0 + r;
            sA[r * 32 + kk] = (cell < NC) ? tqH[(size_t)cell * NBGDIM + ks * 32 + kk]
                                          : (_Float16)0.f;
        }
        __syncthreads();
        if (ks + 1 < 64)
            __builtin_prefetch(WpH + (((size_t)(wv * 64 + ks + 1)) << 9), 0, 1);
        v16h a  = loadAv(sA, 32, lane);
        v16h b0 = loadBs(WpH, 64, ks, wv, lane);
        v16h b1 = loadBs(WpH, 64, ks, wv + 16, lane);
        acc0 = wmma32(a, b0, acc0);
        acc1 = wmma32(a, b1, acc1);
        __syncthreads();
    }
#pragma unroll
    for (int r = 0; r < 8; ++r) {
        int M = r + ((lane >> 4) << 3);
        int cell = row0 + M;
        if (cell < NC) {
            int c0 = wv * 16 + (lane & 15), c1 = 256 + wv * 16 + (lane & 15);
            float o0 = acc0[r] + bp[c0];
            float o1 = acc1[r] + bp[c1];
            if (counts[cell] == 0) { o0 = 0.f; o1 = 0.f; }
            out[(size_t)cell * CL + c0] = o0;
            out[(size_t)cell * CL + c1] = o1;
        }
    }
}

// ---------------- launch ----------------
extern "C" void kernel_launch(void* const* d_in, const int* in_sizes, int n_in,
                              void* d_out, int out_size, void* d_ws, size_t ws_size,
                              hipStream_t stream) {
    const float* x    = (const float*)d_in[0];
    const int*   li   = (const int*)d_in[1];
    const float* bg   = (const float*)d_in[2];
    const float* ln1g = (const float*)d_in[3];
    const float* ln1b = (const float*)d_in[4];
    const float* lncg = (const float*)d_in[5];
    const float* lncb = (const float*)d_in[6];
    const float* Wq   = (const float*)d_in[7];
    const float* Wk   = (const float*)d_in[8];
    const float* Wv   = (const float*)d_in[9];
    const float* Wo   = (const float*)d_in[10];
    const float* ln2g = (const float*)d_in[11];
    const float* ln2b = (const float*)d_in[12];
    const float* Wf1  = (const float*)d_in[13];
    const float* bf1  = (const float*)d_in[14];
    const float* Wf2  = (const float*)d_in[15];
    const float* bf2  = (const float*)d_in[16];
    const float* Wp   = (const float*)d_in[17];
    const float* bp   = (const float*)d_in[18];

    // workspace carve-out
    char* w = (char*)d_ws;
    auto carve = [&](size_t bytes) {
        char* p = w;
        w += (bytes + 255) & ~(size_t)255;
        return p;
    };
    int*       counts = (int*)      carve((size_t)NC * 4);
    _Float16*  nctx   = (_Float16*) carve((size_t)NC * LPAD * DIM * 2);
    _Float16*  tqH    = (_Float16*) carve((size_t)NC * NBGDIM * 2);
    _Float16*  WqH    = (_Float16*) carve((size_t)DEPTH * DIM * DIM * 2);
    _Float16*  WkH    = (_Float16*) carve((size_t)DEPTH * DIM * DIM * 2);
    _Float16*  WvH    = (_Float16*) carve((size_t)DEPTH * DIM * DIM * 2);
    _Float16*  WoH    = (_Float16*) carve((size_t)DEPTH * DIM * DIM * 2);
    _Float16*  Wf1H   = (_Float16*) carve((size_t)DEPTH * DIM * FF * 2);
    _Float16*  Wf2H   = (_Float16*) carve((size_t)DEPTH * FF * DIM * 2);
    _Float16*  WpH    = (_Float16*) carve((size_t)NBGDIM * CL * 2);

    hipMemsetAsync(counts, 0, (size_t)NC * 4, stream);
    hipMemsetAsync(nctx, 0, (size_t)NC * LPAD * DIM * 2, stream);

    const int nSmall = DEPTH * DIM * DIM;   // 524288
    const int nBig   = DEPTH * DIM * FF;    // 2097152
    const int nWp    = NBGDIM * CL;         // 1048576
    // swizzle-convert all weights into B-fragment order
    swz_kernel<<<(nSmall + 255) / 256, 256, 0, stream>>>(Wq, WqH, DIM, DIM, DEPTH);
    swz_kernel<<<(nSmall + 255) / 256, 256, 0, stream>>>(Wk, WkH, DIM, DIM, DEPTH);
    swz_kernel<<<(nSmall + 255) / 256, 256, 0, stream>>>(Wv, WvH, DIM, DIM, DEPTH);
    swz_kernel<<<(nSmall + 255) / 256, 256, 0, stream>>>(Wo, WoH, DIM, DIM, DEPTH);
    swz_kernel<<<(nBig + 255) / 256, 256, 0, stream>>>(Wf1, Wf1H, DIM, FF, DEPTH);
    swz_kernel<<<(nBig + 255) / 256, 256, 0, stream>>>(Wf2, Wf2H, FF, DIM, DEPTH);
    swz_kernel<<<(nWp + 255) / 256, 256, 0, stream>>>(Wp, WpH, NBGDIM, CL, 1);

    bucket_norm_kernel<<<(NOBS + 7) / 8, 256, 0, stream>>>(x, li, counts, nctx);

    obs_cross_tr_kernel<<<NC, 512, 0, stream>>>(
        counts, nctx, bg, ln1g, ln1b, lncg, lncb,
        WqH, WkH, WvH, WoH, ln2g, ln2b, Wf1H, bf1, Wf2H, bf2, tqH);

    final_proj_kernel<<<(NC + 15) / 16, 512, 0, stream>>>(
        tqH, WpH, bp, counts, (float*)d_out);
}
